// DynamicGCN5_depr_36197984370761
// MI455X (gfx1250) — matrix operand order repaired and checked by reading, not compiled, not run on hardware
//
#include <hip/hip_runtime.h>
#include <hip/hip_bf16.h>
#include <stdint.h>

// ---------------------------------------------------------------------------
// MI455X (gfx1250) fused NNConv GNN.
// Dominant cost: per-layer dense GEMM chain over E=160000 edges:
//   H = relu(edge_attr[16x8] @ ew1^T[8x128]) ; W = H @ ew2^T[128 x d^2]
// fused per 16-edge tile with v_wmma_f32_16x16x32_f16 (fp32 accum), never
// materializing the [E, d^2] tensor (saves ~2.6 GB of HBM traffic).
// ---------------------------------------------------------------------------

typedef _Float16 v16h __attribute__((ext_vector_type(16)));
typedef float    v8f  __attribute__((ext_vector_type(8)));

union Frag { uint32_t u[8]; v16h v; };
union H2   { uint32_t u; _Float16 h[2]; };

__device__ __forceinline__ uint32_t pack_f16(float lo, float hi) {
  H2 t; t.h[0] = (_Float16)lo; t.h[1] = (_Float16)hi; return t.u;
}

// ---------------- small utility kernels ----------------

__global__ void fill_f32(float* __restrict__ p, float v, int n) {
  int i = blockIdx.x * blockDim.x + threadIdx.x;
  if (i < n) p[i] = v;
}

__global__ void deg_count(const int* __restrict__ col, float* __restrict__ deg, int E) {
  int e = blockIdx.x * blockDim.x + threadIdx.x;
  if (e < E) atomicAdd(&deg[col[e]], 1.0f);
}

// x_proj[n,j] = sum_i x[n,i]*pw[j,i] + pb[j]
__global__ void node_proj(const float* __restrict__ x, const float* __restrict__ pw,
                          const float* __restrict__ pb, float* __restrict__ xp,
                          int n_nodes, int din, int dout) {
  int t = blockIdx.x * blockDim.x + threadIdx.x;
  if (t >= n_nodes * dout) return;
  int n = t / dout, j = t - n * dout;
  const float* xr = x + n * din;
  const float* wr = pw + j * din;
  float acc = pb[j];
  for (int i = 0; i < din; ++i) acc += xr[i] * wr[i];
  xp[t] = acc;
}

// x_new = relu(agg/deg + xp@rw^T + cb) + xp
__global__ void node_update(const float* __restrict__ xp, const float* __restrict__ agg,
                            const float* __restrict__ deg, const float* __restrict__ rw,
                            const float* __restrict__ cb, float* __restrict__ xout,
                            int n_nodes, int D) {
  int t = blockIdx.x * blockDim.x + threadIdx.x;
  if (t >= n_nodes * D) return;
  int n = t / D, j = t - n * D;
  float dg = deg[n]; dg = dg < 1.0f ? 1.0f : dg;
  float acc = agg[t] / dg + cb[j];
  const float* xr = xp + n * D;
  const float* wr = rw + j * D;
  for (int i = 0; i < D; ++i) acc += xr[i] * wr[i];
  float r = acc > 0.0f ? acc : 0.0f;
  xout[t] = r + xp[t];
}

// Final per-edge MLP: 16 -> 16 -> 1
__global__ void edge_mlp(const float* __restrict__ x, const int* __restrict__ row,
                         const int* __restrict__ col,
                         const float* __restrict__ w0, const float* __restrict__ b0,
                         const float* __restrict__ w1, const float* __restrict__ b1,
                         float* __restrict__ out, int E) {
  int e = blockIdx.x * blockDim.x + threadIdx.x;
  if (e >= E) return;
  const float* xr = x + row[e] * 16;
  const float* xc = x + col[e] * 16;
  float er[16];
  #pragma unroll
  for (int i = 0; i < 16; ++i) er[i] = xr[i] + xc[i];
  float o = b1[0];
  #pragma unroll
  for (int h = 0; h < 16; ++h) {
    float a = b0[h];
    #pragma unroll
    for (int i = 0; i < 16; ++i) a += er[i] * w0[h * 16 + i];
    a = a > 0.0f ? a : 0.0f;
    o += a * w1[h];
  }
  out[e] = o;
}

// ---------------- WMMA B-operand repack kernels ----------------
// B layout for 16x16x32 f16 (32x16 K x N), wave32 (derived from ISA 7.12.4/5):
//   lane L: N = L%16 ;  K = 16*(v/4) + 8*(L/16) + 2*(v%4) + half  for VGPR v in [0,8)
// Packed buffer order: [ntile][lane][v] so each lane streams 8 contiguous dwords.

// ew1: [128,8].  B1[k,n] = ew1[n,k], k padded 8->32 with zeros.  8 n-tiles.
__global__ void pack_b1(const float* __restrict__ ew1, uint32_t* __restrict__ b1p) {
  int idx = blockIdx.x * blockDim.x + threadIdx.x;   // 8*32*8 = 2048
  if (idx >= 8 * 32 * 8) return;
  int v = idx & 7, lane = (idx >> 3) & 31, nt = idx >> 8;
  int n = nt * 16 + (lane & 15);
  int k = 16 * (v >> 2) + 8 * (lane >> 4) + 2 * (v & 3);
  float lo = (k     < 8) ? ew1[n * 8 + k]     : 0.0f;
  float hi = (k + 1 < 8) ? ew1[n * 8 + k + 1] : 0.0f;
  b1p[idx] = pack_f16(lo, hi);
}

// ew2: [d^2, 128].  B2[k,c] = ew2[c*128+k].  [ntile][kchunk(4)][lane][v]
__global__ void pack_b2(const float* __restrict__ ew2, uint32_t* __restrict__ b2p, int ntiles) {
  int idx = blockIdx.x * blockDim.x + threadIdx.x;   // ntiles*4*32*8
  if (idx >= ntiles * 1024) return;
  int v = idx & 7, lane = (idx >> 3) & 31, kc = (idx >> 8) & 3, nt = idx >> 10;
  int c = nt * 16 + (lane & 15);
  int k = kc * 32 + 16 * (v >> 2) + 8 * (lane >> 4) + 2 * (v & 3);
  b2p[idx] = pack_f16(ew2[c * 128 + k], ew2[c * 128 + k + 1]);
}

// ---------------- fused edge-conditioned conv kernel ----------------
// Per 16-edge tile: H=relu(A1*B1+eb1) [16x128] -> W tiles = H*B2+eb2 ->
// msg[e,o] += xp[e, c/D] * W[e,c] (LDS atomics) -> scatter-add to agg.

template<int D>
__global__ __launch_bounds__(128)
void edge_conv(const float* __restrict__ edge_attr,      // [E,8]
               const int* __restrict__ row, const int* __restrict__ col,
               const float* __restrict__ xp,             // [N,D]
               const uint32_t* __restrict__ b1p,         // [8][32][8]
               const float* __restrict__ eb1,            // [128]
               const uint32_t* __restrict__ b2p,         // [D*D/16][4][32][8]
               const float* __restrict__ eb2,            // [D*D]
               float* __restrict__ agg,                  // [N,D]
               int E) {
  constexpr int NT2 = (D * D) / 16;
  __shared__ uint32_t sA1[32][8];    // GEMM1 A fragments (f16 pairs), shared by all waves
  __shared__ uint32_t sH[4][32][8];  // GEMM2 A fragments: H in per-kchunk A layout
  __shared__ float    sXP[16][D];
  __shared__ float    sMsg[16][D];
  __shared__ int      sRow[16], sCol[16];

  const int e0   = blockIdx.x * 16;
  const int tid  = threadIdx.x;
  const int lane = tid & 31;
  const int wave = tid >> 5;

  if (tid < 16) {
    int e = e0 + tid;
    sRow[tid] = (e < E) ? row[e] : 0;
    sCol[tid] = (e < E) ? col[e] : -1;
  }
  __syncthreads();

  for (int i = tid; i < 16 * D; i += 128) {
    int e = i / D, j = i - e * D;
    sXP[e][j]  = xp[sRow[e] * D + j];
    sMsg[e][j] = 0.0f;
  }
  // Build A1 (16x32, K padded 8->32). A layout (ISA 7.12.2, 16-bit A 16x32):
  //   lane L: M = L%16 ; VGPR v: K = 16*(v/4) + 8*(L/16) + 2*(v%4) + half
  for (int i = tid; i < 32 * 8; i += 128) {
    int l = i >> 3, v = i & 7;
    int m = l & 15;
    int k = ((v >> 2) << 4) + ((l >> 4) << 3) + ((v & 3) << 1);
    int e = e0 + m;
    float lo = (k     < 8 && e < E) ? edge_attr[e * 8 + k]     : 0.0f;
    float hi = (k + 1 < 8 && e < E) ? edge_attr[e * 8 + k + 1] : 0.0f;
    sA1[l][v] = pack_f16(lo, hi);
  }
  __syncthreads();

  // ---- GEMM1: H = relu(A1 * B1 + eb1); 8 n-tiles, 2 per wave ----
  Frag a1;
  #pragma unroll
  for (int v = 0; v < 8; ++v) a1.u[v] = sA1[lane][v];

  for (int nt = wave; nt < 8; nt += 4) {
    Frag b;
    const uint32_t* bp = b1p + (nt * 32 + lane) * 8;
    #pragma unroll
    for (int v = 0; v < 8; ++v) b.u[v] = bp[v];
    v8f acc = {};
    acc = __builtin_amdgcn_wmma_f32_16x16x32_f16(false, a1.v, false, b.v,
                                                 (short)0, acc, false, false);
    // C/D layout: lane L -> col n = nt*16 + L%16 ; reg r -> row m = r + 8*(L/16)
    int kcol = nt * 16 + (lane & 15);     // this is the K index of H for GEMM2
    float bias = eb1[kcol];
    int mBase = (lane >> 4) * 8;
    int kc = kcol >> 5, kl = kcol & 31;
    int hi16 = kl >> 4, rem = kl & 15, g = rem >> 3, rem2 = rem & 7;
    int vD = (hi16 << 2) + (rem2 >> 1), half = rem2 & 1;
    #pragma unroll
    for (int r = 0; r < 8; ++r) {
      float h = acc[r] + bias;
      h = h > 0.0f ? h : 0.0f;
      int laneD = (mBase + r) + (g << 4);
      ((_Float16*)&sH[kc][laneD][vD])[half] = (_Float16)h;   // 16-bit DS store
    }
  }
  __syncthreads();

  // ---- GEMM2: W tile = H * B2 + eb2, consumed in-register ----
  Frag a2[4];
  #pragma unroll
  for (int kc = 0; kc < 4; ++kc)
    #pragma unroll
    for (int v = 0; v < 8; ++v) a2[kc].u[v] = sH[kc][lane][v];

  for (int nt = wave; nt < NT2; nt += 4) {
    v8f acc = {};
    #pragma unroll
    for (int kc = 0; kc < 4; ++kc) {
      Frag b;
      const uint32_t* bp = b2p + ((nt * 4 + kc) * 32 + lane) * 8;
      #pragma unroll
      for (int v = 0; v < 8; ++v) b.u[v] = bp[v];
      acc = __builtin_amdgcn_wmma_f32_16x16x32_f16(false, a2[kc].v, false, b.v,
                                                   (short)0, acc, false, false);
    }
    int c = nt * 16 + (lane & 15);      // flat (i,o): c = i*D + o
    float bias = eb2[c];
    int i = c / D, o = c & (D - 1);     // D is a power of two
    int mBase = (lane >> 4) * 8;
    #pragma unroll
    for (int r = 0; r < 8; ++r) {
      int e = mBase + r;
      float w = acc[r] + bias;
      atomicAdd(&sMsg[e][o], sXP[e][i] * w);   // ds_add_f32
    }
  }
  __syncthreads();

  for (int idx = tid; idx < 16 * D; idx += 128) {
    int e = idx / D, o = idx - e * D;
    int c = sCol[e];
    if (c >= 0) atomicAdd(&agg[c * D + o], sMsg[e][o]);   // global_atomic_add_f32
  }
}

// ---------------- orchestration ----------------

extern "C" void kernel_launch(void* const* d_in, const int* in_sizes, int n_in,
                              void* d_out, int out_size, void* d_ws, size_t ws_size,
                              hipStream_t stream) {
  const int N = 10000, E = 160000;
  const float* x_in      = (const float*)d_in[0];
  const float* edge_attr = (const float*)d_in[1];
  const int*   edge_idx  = (const int*)d_in[2];
  const int*   row = edge_idx;
  const int*   col = edge_idx + E;
  auto P = [&](int L, int k) -> const float* { return (const float*)d_in[3 + L * 8 + k]; };
  const float* m_w0 = (const float*)d_in[35];
  const float* m_b0 = (const float*)d_in[36];
  const float* m_w1 = (const float*)d_in[37];
  const float* m_b1 = (const float*)d_in[38];

  float* ws  = (float*)d_ws;
  float* xp  = ws;  ws += N * 32;
  float* xA  = ws;  ws += N * 32;
  float* xB  = ws;  ws += N * 32;
  float* agg = ws;  ws += N * 32;
  float* deg = ws;  ws += N;
  uint32_t* b1p = (uint32_t*)ws;  ws += 8 * 32 * 8;
  uint32_t* b2p = (uint32_t*)ws;  ws += 64 * 4 * 32 * 8;

  fill_f32<<<(N + 255) / 256, 256, 0, stream>>>(deg, 0.0f, N);
  deg_count<<<(E + 255) / 256, 256, 0, stream>>>(col, deg, E);

  const int DOUT[4] = {32, 16, 16, 16};
  int din = 16;
  const float* xcur = x_in;
  float* xbufs[2] = {xA, xB};
  for (int L = 0; L < 4; ++L) {
    const int D = DOUT[L];
    const int nt2 = D * D / 16;
    node_proj<<<(N * D + 255) / 256, 256, 0, stream>>>(xcur, P(L, 0), P(L, 1), xp, N, din, D);
    fill_f32<<<(N * D + 255) / 256, 256, 0, stream>>>(agg, 0.0f, N * D);
    pack_b1<<<(2048 + 255) / 256, 256, 0, stream>>>(P(L, 2), b1p);
    pack_b2<<<(nt2 * 1024 + 255) / 256, 256, 0, stream>>>(P(L, 4), b2p, nt2);
    if (D == 32)
      edge_conv<32><<<(E + 15) / 16, 128, 0, stream>>>(edge_attr, row, col, xp, b1p,
                                                       P(L, 3), b2p, P(L, 5), agg, E);
    else
      edge_conv<16><<<(E + 15) / 16, 128, 0, stream>>>(edge_attr, row, col, xp, b1p,
                                                       P(L, 3), b2p, P(L, 5), agg, E);
    float* xnext = xbufs[L & 1];
    node_update<<<(N * D + 255) / 256, 256, 0, stream>>>(xp, agg, deg, P(L, 6), P(L, 7),
                                                         xnext, N, D);
    xcur = xnext;
    din = D;
  }
  edge_mlp<<<(E + 255) / 256, 256, 0, stream>>>(xcur, row, col, m_w0, m_b0, m_w1, m_b1,
                                                (float*)d_out, E);
}
